// MultiHeadAttention_83476984365531
// MI455X (gfx1250) — matrix operand order repaired and checked by reading, not compiled, not run on hardware
//
#include <hip/hip_runtime.h>
#include <hip/hip_bf16.h>

typedef __bf16 bf16;
typedef bf16  v8bf  __attribute__((ext_vector_type(8)));
typedef bf16  v16bf __attribute__((ext_vector_type(16)));
typedef float v8f   __attribute__((ext_vector_type(8)));

#define WMMA_BF16(a, b, c) \
  __builtin_amdgcn_wmma_f32_16x16x32_bf16(false, (a), false, (b), (short)0, (c), false, false)

// Problem dimensions (match the reference)
constexpr int BATCH = 8;
constexpr int SEQ   = 1024;   // T
constexpr int DM    = 1024;   // d_model
constexpr int NH    = 16;     // heads
constexpr int DHD   = 64;     // d_head
constexpr int HD    = NH * DHD;       // 1024
constexpr int BT    = BATCH * SEQ;    // 8192

// ---------------- fragment load helpers (contiguous k order, k = 16*hi + j) ----------------
// A fragment (16x32): lane m = lane&15, hi = lane>>4.
// elements 0..7  -> K = k0 + 16*hi + j      (16B contiguous)
// elements 8..15 -> K = k0 + 16*hi + 8 + j  (16B contiguous)
__device__ __forceinline__ v16bf load_a16(const bf16* __restrict__ row, int k0, int hi) {
  v8bf lo = *(const v8bf*)(row + k0 + 16 * hi);
  v8bf up = *(const v8bf*)(row + k0 + 16 * hi + 8);
  return __builtin_shufflevector(lo, up, 0,1,2,3,4,5,6,7,8,9,10,11,12,13,14,15);
}
// B fragment (32x16), fed from N-major (K-contiguous) storage: lane n = lane&15.
// elements j -> K = k0 + 16*hi + j (32B contiguous)
__device__ __forceinline__ v16bf load_b16(const bf16* __restrict__ row, int k0, int hi) {
  return *(const v16bf*)(row + k0 + 16 * hi);
}

// ---------------- stage 0: conversion / weight transposes ----------------
__global__ void cvt_f32_to_bf16(const float* __restrict__ in, bf16* __restrict__ out, int n) {
  int i = blockIdx.x * blockDim.x + threadIdx.x;
  if (i < n) out[i] = (bf16)in[i];
}

// W [H, D, DH] fp32 -> Wt [H, DH, D] bf16 (N-major for B operand)
__global__ void transpose_hdk(const float* __restrict__ in, bf16* __restrict__ out) {
  int i = blockIdx.x * blockDim.x + threadIdx.x;  // < NH*DM*DHD
  int h   = i / (DM * DHD);
  int rem = i - h * (DM * DHD);
  int d   = rem / DHD;
  int k   = rem - d * DHD;
  out[(size_t)h * DHD * DM + (size_t)k * DM + d] = (bf16)in[i];
}

// Wo [HD, D] fp32 -> Wot [D, HD] bf16
__global__ void transpose_2d(const float* __restrict__ in, bf16* __restrict__ out) {
  int i = blockIdx.x * blockDim.x + threadIdx.x;  // < HD*DM
  int r = i / DM;
  int c = i - r * DM;
  out[(size_t)c * HD + r] = (bf16)in[i];
}

// ---------------- stage 1: QKV projections ----------------
// Each wave: one 16-row token tile x one head -> 16x64 output. K-loop over DM.
// transposeOut=0: out[B,H,T,DH]; transposeOut=1: out[B,H,DH,T] (for V).
__global__ void proj_kernel(const bf16* __restrict__ X, const bf16* __restrict__ Wt,
                            bf16* __restrict__ Out, int transposeOut) {
  const int lane = threadIdx.x & 31;
  const int wave = threadIdx.x >> 5;
  const int tile = blockIdx.x * 4 + wave;   // 0..511 token tiles
  const int h    = blockIdx.y;
  const int n    = lane & 15;
  const int hi   = lane >> 4;
  const int rt0  = tile * 16;
  const int b    = rt0 / SEQ;
  const int t0   = rt0 - b * SEQ;

  const bf16* arow  = X + (size_t)(rt0 + n) * DM;      // A: lane's row m = n
  const bf16* wbase = Wt + (size_t)h * DHD * DM;

  v8f acc[4];
#pragma unroll
  for (int i = 0; i < 4; i++)
#pragma unroll
    for (int r = 0; r < 8; r++) acc[i][r] = 0.0f;

  for (int k0 = 0; k0 < DM; k0 += 32) {
    v16bf a = load_a16(arow, k0, hi);
#pragma unroll
    for (int nb = 0; nb < 4; nb++) {
      const bf16* brow = wbase + (size_t)(nb * 16 + n) * DM;
      acc[nb] = WMMA_BF16(a, load_b16(brow, k0, hi), acc[nb]);
    }
  }

  if (!transposeOut) {
    bf16* obase = Out + ((size_t)(b * NH + h) * SEQ + t0) * DHD;
#pragma unroll
    for (int nb = 0; nb < 4; nb++)
#pragma unroll
      for (int r = 0; r < 8; r++)
        obase[(size_t)(r + 8 * hi) * DHD + nb * 16 + n] = (bf16)acc[nb][r];
  } else {
    bf16* obase = Out + (size_t)(b * NH + h) * DHD * SEQ;
#pragma unroll
    for (int nb = 0; nb < 4; nb++) {
      v8bf pk;
#pragma unroll
      for (int r = 0; r < 8; r++) pk[r] = (bf16)acc[nb][r];
      *(v8bf*)(obase + (size_t)(nb * 16 + n) * SEQ + t0 + 8 * hi) = pk;
    }
  }
}

// ---------------- stage 2: causal flash attention ----------------
// One wave per (bh, 16-query tile); 32 keys per step; online softmax.
__global__ void attn_kernel(const bf16* __restrict__ Q, const bf16* __restrict__ K,
                            const bf16* __restrict__ Vt, bf16* __restrict__ AOut) {
  __shared__ bf16 plds[16 * 32];  // P tile, row-major 16x32
  const int lane  = threadIdx.x;
  const int n     = lane & 15;
  const int hi    = lane >> 4;
  const int qt    = blockIdx.x;
  const int bh    = blockIdx.y;       // b*NH + h
  const int qbase = qt * 16;

  const bf16* qrow = Q + ((size_t)bh * SEQ + qbase + n) * DHD;
  v16bf a0 = load_a16(qrow, 0, hi);    // dh 0..31
  v16bf a1 = load_a16(qrow, 32, hi);   // dh 32..63

  float rm[8], rs[8];
  v8f o[4];
#pragma unroll
  for (int r = 0; r < 8; r++) { rm[r] = -3.0e38f; rs[r] = 0.0f; }
#pragma unroll
  for (int i = 0; i < 4; i++)
#pragma unroll
    for (int r = 0; r < 8; r++) o[i][r] = 0.0f;

  const bf16* kbh = K + (size_t)bh * SEQ * DHD;
  const bf16* vbh = Vt + (size_t)bh * DHD * SEQ;
  const int nsteps = (qbase + 47) / 32;  // keys 0..qbase+15 covered

  for (int s = 0; s < nsteps; s++) {
    const int kb = s * 32;
    const bf16* krow0 = kbh + (size_t)(kb + n) * DHD;
    const bf16* krow1 = kbh + (size_t)(kb + 16 + n) * DHD;

    v8f s0, s1;
#pragma unroll
    for (int r = 0; r < 8; r++) { s0[r] = 0.0f; s1[r] = 0.0f; }
    s0 = WMMA_BF16(a0, load_b16(krow0, 0, hi), s0);
    s0 = WMMA_BF16(a1, load_b16(krow0, 32, hi), s0);
    s1 = WMMA_BF16(a0, load_b16(krow1, 0, hi), s1);
    s1 = WMMA_BF16(a1, load_b16(krow1, 32, hi), s1);

    float p0[8], p1[8];
#pragma unroll
    for (int r = 0; r < 8; r++) {
      const int qg = qbase + r + 8 * hi;
      float v0 = s0[r] * 0.125f;               // DH^-0.5
      float v1 = s1[r] * 0.125f;
      if (kb + n > qg)      v0 = -3.0e38f;     // causal mask
      if (kb + 16 + n > qg) v1 = -3.0e38f;
      float mx = fmaxf(v0, v1);
#pragma unroll
      for (int msk = 1; msk < 16; msk <<= 1) mx = fmaxf(mx, __shfl_xor(mx, msk, 32));
      const float mnew  = fmaxf(rm[r], mx);
      const float alpha = __expf(rm[r] - mnew);
      rm[r] = mnew;
      const float e0 = __expf(v0 - mnew);
      const float e1 = __expf(v1 - mnew);
      float psum = e0 + e1;
#pragma unroll
      for (int msk = 1; msk < 16; msk <<= 1) psum += __shfl_xor(psum, msk, 32);
      rs[r] = rs[r] * alpha + psum;
#pragma unroll
      for (int nb = 0; nb < 4; nb++) o[nb][r] *= alpha;
      p0[r] = e0;
      p1[r] = e1;
    }

    // C-layout -> A-layout transpose of P through LDS
#pragma unroll
    for (int r = 0; r < 8; r++) {
      const int m = r + 8 * hi;
      plds[m * 32 + n]      = (bf16)p0[r];
      plds[m * 32 + 16 + n] = (bf16)p1[r];
    }
    __syncthreads();
    v16bf pa = load_a16(plds + n * 32, 0, hi);  // full K=32 (32 keys)
#pragma unroll
    for (int nb = 0; nb < 4; nb++) {
      const bf16* vrow = vbh + (size_t)(nb * 16 + n) * SEQ + kb;
      o[nb] = WMMA_BF16(pa, load_b16(vrow, 0, hi), o[nb]);
    }
    __syncthreads();
  }

  // normalize and store head-concat layout [B, T, H*DH] as bf16
  const int b = bh / NH, h = bh - b * NH;
  bf16* obase = AOut + ((size_t)b * SEQ + qbase) * HD + h * DHD;
#pragma unroll
  for (int r = 0; r < 8; r++) {
    const float inv = 1.0f / rs[r];
    const int m = r + 8 * hi;
#pragma unroll
    for (int nb = 0; nb < 4; nb++)
      obase[(size_t)m * HD + nb * 16 + n] = (bf16)(o[nb][r] * inv);
  }
}

// ---------------- stage 3: output projection + bias ----------------
__global__ void outproj_kernel(const bf16* __restrict__ A, const bf16* __restrict__ Wot,
                               const float* __restrict__ bo, float* __restrict__ Out) {
  const int lane = threadIdx.x & 31;
  const int wave = threadIdx.x >> 5;
  const int tile = blockIdx.x * 4 + wave;     // 0..511
  const int ncb  = blockIdx.y * 64;
  const int n    = lane & 15;
  const int hi   = lane >> 4;

  const bf16* arow = A + (size_t)(tile * 16 + n) * HD;

  v8f acc[4];
#pragma unroll
  for (int i = 0; i < 4; i++)
#pragma unroll
    for (int r = 0; r < 8; r++) acc[i][r] = 0.0f;

  for (int k0 = 0; k0 < HD; k0 += 32) {
    v16bf a = load_a16(arow, k0, hi);
#pragma unroll
    for (int nb = 0; nb < 4; nb++) {
      const bf16* brow = Wot + (size_t)(ncb + nb * 16 + n) * HD;
      acc[nb] = WMMA_BF16(a, load_b16(brow, k0, hi), acc[nb]);
    }
  }

#pragma unroll
  for (int nb = 0; nb < 4; nb++) {
    const int col = ncb + nb * 16 + n;
    const float bias = bo[col];
#pragma unroll
    for (int r = 0; r < 8; r++)
      Out[(size_t)(tile * 16 + r + 8 * hi) * DM + col] = acc[nb][r] + bias;
  }
}

// ---------------- launcher ----------------
extern "C" void kernel_launch(void* const* d_in, const int* in_sizes, int n_in,
                              void* d_out, int out_size, void* d_ws, size_t ws_size,
                              hipStream_t stream) {
  const float* x  = (const float*)d_in[0];
  const float* Wq = (const float*)d_in[1];
  const float* Wk = (const float*)d_in[2];
  const float* Wv = (const float*)d_in[3];
  const float* Wo = (const float*)d_in[4];
  const float* bo = (const float*)d_in[5];
  float* out = (float*)d_out;

  char* ws = (char*)d_ws;
  size_t off = 0;
  bf16* xb  = (bf16*)(ws + off); off += (size_t)BT * DM * sizeof(bf16);        // 16 MB
  bf16* Wqt = (bf16*)(ws + off); off += (size_t)NH * DHD * DM * sizeof(bf16);  //  2 MB
  bf16* Wkt = (bf16*)(ws + off); off += (size_t)NH * DHD * DM * sizeof(bf16);
  bf16* Wvt = (bf16*)(ws + off); off += (size_t)NH * DHD * DM * sizeof(bf16);
  bf16* Wot = (bf16*)(ws + off); off += (size_t)DM * HD * sizeof(bf16);
  bf16* Qb  = (bf16*)(ws + off); off += (size_t)BATCH * NH * SEQ * DHD * sizeof(bf16);  // 16 MB
  bf16* Kb  = (bf16*)(ws + off); off += (size_t)BATCH * NH * SEQ * DHD * sizeof(bf16);
  bf16* Vtb = (bf16*)(ws + off); off += (size_t)BATCH * NH * DHD * SEQ * sizeof(bf16);
  bf16* Ab  = (bf16*)(ws + off); off += (size_t)BT * HD * sizeof(bf16);        // 16 MB

  // stage 0: conversions
  {
    const int n = BT * DM;
    cvt_f32_to_bf16<<<n / 256, 256, 0, stream>>>(x, xb, n);
    const int nw = NH * DM * DHD;
    transpose_hdk<<<nw / 256, 256, 0, stream>>>(Wq, Wqt);
    transpose_hdk<<<nw / 256, 256, 0, stream>>>(Wk, Wkt);
    transpose_hdk<<<nw / 256, 256, 0, stream>>>(Wv, Wvt);
    transpose_2d<<<(HD * DM) / 256, 256, 0, stream>>>(Wo, Wot);
  }

  // stage 1: projections (512 token tiles / 4 waves per block)
  {
    dim3 grid(BT / 16 / 4, NH);
    proj_kernel<<<grid, 128, 0, stream>>>(xb, Wqt, Qb, 0);
    proj_kernel<<<grid, 128, 0, stream>>>(xb, Wkt, Kb, 0);
    proj_kernel<<<grid, 128, 0, stream>>>(xb, Wvt, Vtb, 1);  // V stored transposed
  }

  // stage 2: flash attention, one wave per (bh, q-tile)
  {
    dim3 grid(SEQ / 16, BATCH * NH);
    attn_kernel<<<grid, 32, 0, stream>>>(Qb, Kb, Vtb, Ab);
  }

  // stage 3: output projection + bias
  {
    dim3 grid(BT / 16 / 4, DM / 64);
    outproj_kernel<<<grid, 128, 0, stream>>>(Ab, Wot, bo, out);
  }
}
